// MultiheadAttention_2336462209740
// MI455X (gfx1250) — compile-verified, matrix operand order checked by reading, not run on hardware
//
#include <hip/hip_runtime.h>

// Problem constants (match reference)
#define BB 4
#define SS 2048
#define DD 1024
#define HH 16
#define DKK 64

typedef __attribute__((ext_vector_type(16))) _Float16 v16h;
typedef __attribute__((ext_vector_type(8)))  float    v8f;

union FragU {
    v16h h;
    uint4 q[2];
};

__device__ __forceinline__ v8f wmma_f16(v16h a, v16h b, v8f c) {
    // D = A(16x32 f16) * B(32x16 f16) + C(16x16 f32)
    return __builtin_amdgcn_wmma_f32_16x16x32_f16(
        /*neg_a=*/false, a, /*neg_b=*/false, b,
        /*c_mod=*/(short)0, c, /*reuse_a=*/false, /*reuse_b=*/false);
}

// ---------------------------------------------------------------------------
// One-time f32 -> f16 conversion (8 elements / thread, b128 in / b128 out).
// ---------------------------------------------------------------------------
__global__ __launch_bounds__(256) void cvt_f32_f16(const float* __restrict__ in,
                                                   _Float16* __restrict__ out, int n8) {
    const int i = blockIdx.x * blockDim.x + threadIdx.x;
    if (i >= n8) return;
    const float4 f0 = ((const float4*)in)[2 * i];
    const float4 f1 = ((const float4*)in)[2 * i + 1];
    union { _Float16 h[8]; uint4 u; } r;
    r.h[0] = (_Float16)f0.x; r.h[1] = (_Float16)f0.y;
    r.h[2] = (_Float16)f0.z; r.h[3] = (_Float16)f0.w;
    r.h[4] = (_Float16)f1.x; r.h[5] = (_Float16)f1.y;
    r.h[6] = (_Float16)f1.z; r.h[7] = (_Float16)f1.w;
    ((uint4*)out)[i] = r.u;
}

// ---------------------------------------------------------------------------
// GEMM: Out[m][n] = sum_k A[m][k] * W[n][k]  (A @ W^T), A/W are f16, K=N=1024.
// OMODE 0: f16 row-major out
// OMODE 1: f16 per-head-transposed out Vt[(b*H+h)*64+dk][s]  (attention V)
// OMODE 2: f32 row-major out (final result)
// OMODE 3: f16 row-major out, scaled by 1/sqrt(dk)*log2(e)   (attention Q)
// Block = 256 threads (8 waves). Block tile 256(M) x 64(N); wave tile 32x64.
// Inner loop: 3 pointer bumps, 12 b128 loads (immediate offsets), 8 WMMAs.
// ---------------------------------------------------------------------------
template <int OMODE>
__global__ __launch_bounds__(256) void gemm_xWT(const _Float16* __restrict__ A,
                                                const _Float16* __restrict__ W,
                                                void* __restrict__ Outp) {
    const int lane = threadIdx.x & 31;
    const int wave = threadIdx.x >> 5;
    const int ln   = lane & 15;
    const int hv   = lane >> 4;
    const int hf   = hv * 8;

    const int rowbase = blockIdx.x * 256 + wave * 32;
    const int colbase = blockIdx.y * 64;

    v8f acc[2][4] = {};

    const _Float16* aptr0 = A + (size_t)(rowbase + ln) * DD + hf;
    const _Float16* aptr1 = aptr0 + (size_t)16 * DD;
    const _Float16* wptr  = W + (size_t)(colbase + ln) * DD + hf;

    for (int kb = 0; kb < DD; kb += 32) {
        FragU a0, a1, bf[4];
        a0.q[0] = *(const uint4*)(aptr0);
        a0.q[1] = *(const uint4*)(aptr0 + 16);
        a1.q[0] = *(const uint4*)(aptr1);
        a1.q[1] = *(const uint4*)(aptr1 + 16);
#pragma unroll
        for (int nt = 0; nt < 4; ++nt) {
            bf[nt].q[0] = *(const uint4*)(wptr + nt * 16 * DD);
            bf[nt].q[1] = *(const uint4*)(wptr + nt * 16 * DD + 16);
        }
#pragma unroll
        for (int nt = 0; nt < 4; ++nt) {
            acc[0][nt] = wmma_f16(a0.h, bf[nt].h, acc[0][nt]);
            acc[1][nt] = wmma_f16(a1.h, bf[nt].h, acc[1][nt]);
        }
        aptr0 += 32;
        aptr1 += 32;
        wptr  += 32;
    }

    constexpr float QSCALE = 0.125f * 1.4426950408889634f;  // 1/sqrt(64) * log2(e)

#pragma unroll
    for (int mi = 0; mi < 2; ++mi) {
        if constexpr (OMODE == 0 || OMODE == 3) {
            _Float16* Out = (_Float16*)Outp;
#pragma unroll
            for (int nt = 0; nt < 4; ++nt)
#pragma unroll
                for (int r = 0; r < 8; ++r) {
                    const int row = rowbase + mi * 16 + r + 8 * hv;
                    const int col = colbase + nt * 16 + ln;
                    float vv = acc[mi][nt][r];
                    if constexpr (OMODE == 3) vv *= QSCALE;
                    Out[(size_t)row * DD + col] = (_Float16)vv;
                }
        } else if constexpr (OMODE == 1) {
            _Float16* Out = (_Float16*)Outp;  // Vt[(b*H+h)*64+dk][s]
#pragma unroll
            for (int nt = 0; nt < 4; ++nt)
#pragma unroll
                for (int r = 0; r < 8; ++r) {
                    const int row = rowbase + mi * 16 + r + 8 * hv;  // b*S + s
                    const int b   = row >> 11;
                    const int s   = row & (SS - 1);
                    const int col = colbase + nt * 16 + ln;          // h*64 + dk
                    const int hh  = col >> 6;
                    const int dk  = col & 63;
                    Out[((size_t)((b * HH + hh) * DKK + dk)) * SS + s] = (_Float16)acc[mi][nt][r];
                }
        } else {
            float* Out = (float*)Outp;
#pragma unroll
            for (int nt = 0; nt < 4; ++nt)
#pragma unroll
                for (int r = 0; r < 8; ++r) {
                    const int row = rowbase + mi * 16 + r + 8 * hv;
                    const int col = colbase + nt * 16 + ln;
                    Out[(size_t)row * DD + col] = acc[mi][nt][r];
                }
        }
    }
}

// ---------------------------------------------------------------------------
// Flash attention, causal. One wave per 16 query rows of one (b,h).
// Qp (pre-scaled by 1/sqrt(dk)*log2 e), Kp: f16 [B*S][D]. Vt: f16 per-head ^T.
// kv step = 64: 8 WMMAs for scores + 8 WMMAs for P*V per online-softmax pass.
// Streams use single bumped pointers + immediate offsets; V loads are issued
// before the softmax so their latency hides under the VALU/shuffle phase.
// No block barriers (divergent causal trip counts across waves).
// ---------------------------------------------------------------------------
__global__ __launch_bounds__(256) void attn_kernel(const _Float16* __restrict__ Qp,
                                                   const _Float16* __restrict__ Kp,
                                                   const _Float16* __restrict__ Vt,
                                                   _Float16* __restrict__ ctx) {
    __shared__ alignas(16) unsigned short pl[8][1024];  // per-wave 16x64 f16 P tile

    const int lane = threadIdx.x & 31;
    const int wave = threadIdx.x >> 5;
    const int ln = lane & 15;
    const int hv = lane >> 4;
    const int hf = hv * 8;

    const int w  = blockIdx.x * 8 + wave;   // 8192 waves total
    const int qt = w & 127;
    const int bh = w >> 7;                  // b*H + h
    const int b  = bh >> 4;
    const int h  = bh & 15;
    const int qbase = qt << 4;

    // Q A-fragments for dk [0,32) and [32,64)
    const _Float16* qrow = Qp + ((size_t)(b * SS + qbase + ln)) * DD + h * DKK + hf;
    FragU a0, a1;
    a0.q[0] = *(const uint4*)(qrow);
    a0.q[1] = *(const uint4*)(qrow + 16);
    a1.q[0] = *(const uint4*)(qrow + 32);
    a1.q[1] = *(const uint4*)(qrow + 48);

    float Mr[8], Lr[8], alpha[8];
    v8f O[4] = {};
#pragma unroll
    for (int r = 0; r < 8; ++r) { Mr[r] = -1e30f; Lr[r] = 0.0f; }

    unsigned short* pb = &pl[wave][0];
    const int nsteps = (qbase + 16 + 63) >> 6;

    // bumped stream pointers (immediate offsets cover the n-tiles)
    const _Float16* kptr  = Kp + ((size_t)(b * SS + ln)) * DD + h * DKK + hf;
    const _Float16* vptr  = Vt + ((size_t)(bh * DKK + ln)) * SS + hf;
    const _Float16* pfptr = Kp + ((size_t)(b * SS + 64 + lane)) * DD + h * DKK;

    for (int it = 0; it < nsteps; ++it) {
        const int kvb = it << 6;

        // --- batch all K fragment loads (16 x b128), then 8 score WMMAs ---
        FragU kf[4][2];
#pragma unroll
        for (int nt = 0; nt < 4; ++nt) {
            const _Float16* kp = kptr + nt * 16 * DD;
            kf[nt][0].q[0] = *(const uint4*)(kp);
            kf[nt][0].q[1] = *(const uint4*)(kp + 16);
            kf[nt][1].q[0] = *(const uint4*)(kp + 32);
            kf[nt][1].q[1] = *(const uint4*)(kp + 48);
        }
        v8f sc[4];
#pragma unroll
        for (int nt = 0; nt < 4; ++nt) {
            v8f c = {};
            c = wmma_f16(a0.h, kf[nt][0].h, c);
            c = wmma_f16(a1.h, kf[nt][1].h, c);
            sc[nt] = c;
        }

        // causal mask (scores already in log2 domain, pre-scaled in Q)
        if ((kvb + 63) > qbase) {
#pragma unroll
            for (int nt = 0; nt < 4; ++nt)
#pragma unroll
                for (int r = 0; r < 8; ++r) {
                    const int row = qbase + r + 8 * hv;
                    const int col = kvb + nt * 16 + ln;
                    if (col > row) sc[nt][r] = -1e30f;
                }
        }

        // --- issue V fragment loads now; latency hides under the softmax ---
        FragU vf[4][2];
#pragma unroll
        for (int nt = 0; nt < 4; ++nt) {
            const _Float16* vp = vptr + nt * 16 * SS;
            vf[nt][0].q[0] = *(const uint4*)(vp);
            vf[nt][0].q[1] = *(const uint4*)(vp + 16);
            vf[nt][1].q[0] = *(const uint4*)(vp + 32);
            vf[nt][1].q[1] = *(const uint4*)(vp + 48);
        }

        // --- online softmax: row r+8*hv lives across the 16 lanes of a half ---
#pragma unroll
        for (int r = 0; r < 8; ++r) {
            float t = fmaxf(fmaxf(sc[0][r], sc[1][r]), fmaxf(sc[2][r], sc[3][r]));
            t = fmaxf(t, __shfl_xor(t, 1, 32));
            t = fmaxf(t, __shfl_xor(t, 2, 32));
            t = fmaxf(t, __shfl_xor(t, 4, 32));
            t = fmaxf(t, __shfl_xor(t, 8, 32));
            const float Mn = fmaxf(Mr[r], t);
            const float al = __builtin_amdgcn_exp2f(Mr[r] - Mn);
            const float p0 = __builtin_amdgcn_exp2f(sc[0][r] - Mn);
            const float p1 = __builtin_amdgcn_exp2f(sc[1][r] - Mn);
            const float p2 = __builtin_amdgcn_exp2f(sc[2][r] - Mn);
            const float p3 = __builtin_amdgcn_exp2f(sc[3][r] - Mn);
            float rs = (p0 + p1) + (p2 + p3);
            rs += __shfl_xor(rs, 1, 32);
            rs += __shfl_xor(rs, 2, 32);
            rs += __shfl_xor(rs, 4, 32);
            rs += __shfl_xor(rs, 8, 32);
            Lr[r] = Lr[r] * al + rs;
            Mr[r] = Mn;
            alpha[r] = al;
            sc[0][r] = p0; sc[1][r] = p1; sc[2][r] = p2; sc[3][r] = p3;
        }
#pragma unroll
        for (int nt = 0; nt < 4; ++nt)
#pragma unroll
            for (int r = 0; r < 8; ++r) O[nt][r] *= alpha[r];

        // --- P (D-frag layout) -> LDS -> two A-frags (16x64 f16) ---
#pragma unroll
        for (int r = 0; r < 8; ++r) {
            const int m = r + 8 * hv;
#pragma unroll
            for (int f = 0; f < 4; ++f) {
                const _Float16 ph = (_Float16)sc[f][r];
                pb[m * 64 + f * 16 + ln] = __builtin_bit_cast(unsigned short, ph);
            }
        }
        // wave-local RAW through LDS: DS ops are in-order per wave; fence compiler
        asm volatile("" ::: "memory");
        FragU pf0, pf1;
        const unsigned short* prow = pb + ln * 64;
        pf0.q[0] = *(const uint4*)(prow + hf);
        pf0.q[1] = *(const uint4*)(prow + 16 + hf);
        pf1.q[0] = *(const uint4*)(prow + 32 + hf);
        pf1.q[1] = *(const uint4*)(prow + 48 + hf);

        // --- O += P * V ---
#pragma unroll
        for (int nt = 0; nt < 4; ++nt) {
            O[nt] = wmma_f16(pf0.h, vf[nt][0].h, O[nt]);
            O[nt] = wmma_f16(pf1.h, vf[nt][1].h, O[nt]);
        }

        if (it + 1 < nsteps) {
            __builtin_prefetch(pfptr, 0, 0);
            __builtin_prefetch(pfptr + (size_t)32 * DD, 0, 0);
        }
        kptr  += (size_t)64 * DD;
        pfptr += (size_t)64 * DD;
        vptr  += 64;
    }

    // normalize and write ctx (reassembled heads, f16 row-major)
#pragma unroll
    for (int r = 0; r < 8; ++r) Lr[r] = 1.0f / Lr[r];
#pragma unroll
    for (int nt = 0; nt < 4; ++nt)
#pragma unroll
        for (int r = 0; r < 8; ++r) {
            const size_t row = (size_t)(b * SS + qbase + r + 8 * hv);
            const int col = h * DKK + nt * 16 + ln;
            ctx[row * DD + col] = (_Float16)(O[nt][r] * Lr[r]);
        }
}

extern "C" void kernel_launch(void* const* d_in, const int* in_sizes, int n_in,
                              void* d_out, int out_size, void* d_ws, size_t ws_size,
                              hipStream_t stream) {
    (void)in_sizes; (void)n_in; (void)out_size; (void)ws_size;
    const float* q  = (const float*)d_in[0];
    const float* k  = (const float*)d_in[1];
    const float* v  = (const float*)d_in[2];
    // d_in[3] = mask (unused: causal mask computed analytically)
    const float* Wq = (const float*)d_in[4];
    const float* Wk = (const float*)d_in[5];
    const float* Wv = (const float*)d_in[6];
    const float* Wo = (const float*)d_in[7];
    float* out = (float*)d_out;

    const size_t NE = (size_t)BB * SS * DD;  // 8388608
    const size_t NW = (size_t)DD * DD;       // 1048576
    _Float16* q16  = (_Float16*)d_ws;
    _Float16* k16  = q16 + NE;
    _Float16* v16  = k16 + NE;
    _Float16* Wq16 = v16 + NE;
    _Float16* Wk16 = Wq16 + NW;
    _Float16* Wv16 = Wk16 + NW;
    _Float16* Wo16 = Wv16 + NW;
    _Float16* Qp   = Wo16 + NW;   // [B*S][D], pre-scaled by 1/sqrt(dk)*log2 e
    _Float16* Kp   = Qp + NE;     // [B*S][D]
    _Float16* Vt   = Kp + NE;     // [(b*H+h)*64+dk][s]
    _Float16* Cx   = Vt + NE;     // [B*S][D]

    dim3 blk(256, 1, 1);

    // one-time f32 -> f16 conversions (memory-bound, ~7us of HBM traffic)
    const int n8x = (int)(NE / 8), n8w = (int)(NW / 8);
    cvt_f32_f16<<<dim3(n8x / 256), blk, 0, stream>>>(q, q16, n8x);
    cvt_f32_f16<<<dim3(n8x / 256), blk, 0, stream>>>(k, k16, n8x);
    cvt_f32_f16<<<dim3(n8x / 256), blk, 0, stream>>>(v, v16, n8x);
    cvt_f32_f16<<<dim3(n8w / 256), blk, 0, stream>>>(Wq, Wq16, n8w);
    cvt_f32_f16<<<dim3(n8w / 256), blk, 0, stream>>>(Wk, Wk16, n8w);
    cvt_f32_f16<<<dim3(n8w / 256), blk, 0, stream>>>(Wv, Wv16, n8w);
    cvt_f32_f16<<<dim3(n8w / 256), blk, 0, stream>>>(Wo, Wo16, n8w);

    dim3 ggrid((BB * SS) / 256, DD / 64, 1);  // (32, 16)
    gemm_xWT<3><<<ggrid, blk, 0, stream>>>(q16, Wq16, (void*)Qp);
    gemm_xWT<0><<<ggrid, blk, 0, stream>>>(k16, Wk16, (void*)Kp);
    gemm_xWT<1><<<ggrid, blk, 0, stream>>>(v16, Wv16, (void*)Vt);

    attn_kernel<<<dim3((BB * HH * SS / 16) / 8, 1, 1), blk, 0, stream>>>(Qp, Kp, Vt, Cx);

    gemm_xWT<2><<<ggrid, blk, 0, stream>>>(Cx, Wo16, (void*)out);
}
